// MultiHeadedAttention_44624710205499
// MI455X (gfx1250) — compile-verified, matrix operand order checked by reading, not run on hardware
//
#include <hip/hip_runtime.h>

// ---------------------------------------------------------------------------
// MI455X (gfx1250, wave32) pooled multi-head attention.
//  - All big matmuls on v_wmma_f32_16x16x32_bf16 (fp32 accum).
//  - fp32->bf16 conversion done ONCE per tensor (no conversion in hot loops).
//  - LDS tiles streamed by the Tensor Data Mover (tensor_load_to_lds),
//    double-buffered, synced with s_wait_tensorcnt.
// Workspace (~68 MB): key_bf, val_bf, qbf, kbf, xbf, Vt (bf16, NT*D each),
//                     Wfkt/W0t/Woutt (bf16 D*D), fk (f32 NT*D).
// ---------------------------------------------------------------------------

typedef unsigned int   u32;
typedef unsigned short u16;
typedef __attribute__((ext_vector_type(16))) __bf16 v16bf;
typedef __attribute__((ext_vector_type(8)))  float  v8f;
typedef __attribute__((ext_vector_type(4)))  u32    v4u;
typedef __attribute__((ext_vector_type(8)))  int    v8i;
typedef __attribute__((ext_vector_type(4)))  int    v4i;

#if __has_builtin(__builtin_amdgcn_tensor_load_to_lds)
#define USE_TDM 1
#else
#define USE_TDM 0
#endif

__device__ __forceinline__ u16 f2bfu(float f) {
    u32 x = __float_as_uint(f);
    u32 r = x + 0x7FFFu + ((x >> 16) & 1u);   // round-to-nearest-even
    return (u16)(r >> 16);
}
__device__ __forceinline__ u32 pack2(float a, float b) {
    return (u32)f2bfu(a) | ((u32)f2bfu(b) << 16);
}

struct FragBF { union { v16bf v; u32 d[8]; }; };

// 16x32 bf16 A/B fragment from an LDS row of 32 bf16 (K pairs contiguous).
// ISA 7.12.2: lanes 0-15 hold K {0..7,16..23}, lanes 16-31 hold {8..15,24..31}.
__device__ __forceinline__ void ldsFrag(FragBF& f, const u16* row, int hi) {
    const u32* rp = (const u32*)row;
#pragma unroll
    for (int j = 0; j < 8; ++j) f.d[j] = rp[((j < 4) ? j : (j + 4)) + hi * 4];
}

#if USE_TDM
// Issue a TDM 2D tile load (bf16 elements): dim0 = row length (tile==tensor),
// dim1 = rows, stride0 = elements between rows. D# per CDNA5 ISA.
// This toolchain's builtin takes 6 args (g0..g4 + cpol); g2/g3/g4 cover the
// 3D..5D / gather descriptor groups and are zero for a plain 2D tile.
__device__ __forceinline__ void tdm_load_2d(const u16* gsrc, u32 lds_addr,
                                            u32 dim0, u32 dim1, u32 stride0) {
    unsigned long long ga = (unsigned long long)(size_t)gsrc;
    v4u g0;
    g0[0] = 1u;                                            // count=1, user mode
    g0[1] = lds_addr;                                      // LDS byte address
    g0[2] = (u32)ga;                                       // global_addr[31:0]
    g0[3] = ((u32)(ga >> 32) & 0x1FFFFFFu) | (2u << 30);   // addr[56:32] | type=2
    v8i g1;
    g1[0] = (int)(1u << 16);                               // data_size=1 (2 bytes)
    g1[1] = (int)((dim0 & 0xFFFFu) << 16);                 // tensor_dim0 lo16
    g1[2] = (int)(((dim0 >> 16) & 0xFFFFu) | ((dim1 & 0xFFFFu) << 16)); // d0 hi | d1 lo
    g1[3] = (int)(((dim1 >> 16) & 0xFFFFu) | ((dim0 & 0xFFFFu) << 16)); // d1 hi | tile_dim0
    g1[4] = (int)(dim1 & 0xFFFFu);                         // tile_dim1 (tile_dim2=0)
    g1[5] = (int)stride0;                                  // tensor_dim0_stride lo32
    g1[6] = 0;                                             // stride0 hi16 | dim1_stride lo16
    g1[7] = 0;
    v4i z4 = (v4i){0, 0, 0, 0};
    v8i z8 = (v8i){0, 0, 0, 0, 0, 0, 0, 0};
    __builtin_amdgcn_tensor_load_to_lds(g0, g1, z4, z4, z8, 0);
}
#else
__device__ __forceinline__ void coop_copy_2d(const u16* __restrict__ gsrc,
                                             u16* __restrict__ ldst,
                                             int dim0, int dim1, int stride0, int tid) {
    int half = (dim0 * dim1) >> 1;
    for (int i = tid; i < half; i += 256) {
        int e = i << 1;
        int r = e / dim0, c = e - r * dim0;
        *(u32*)(ldst + e) = *(const u32*)(gsrc + (size_t)r * stride0 + c);
    }
}
#endif

// ---------------------------------------------------------------------------
// fp32 -> bf16 elementwise (4 per thread, packed dword stores)
// ---------------------------------------------------------------------------
__global__ __launch_bounds__(256) void f32_to_bf16_kernel(const float* __restrict__ in,
                                                          u16* __restrict__ out, int n) {
    int i = (blockIdx.x * 256 + threadIdx.x) * 4;
    if (i + 3 < n) {
        float4 f = *(const float4*)(in + i);
        *(u32*)(out + i)     = pack2(f.x, f.y);
        *(u32*)(out + i + 2) = pack2(f.z, f.w);
    }
}

// W[K][N] fp32 -> Wt[N][K] bf16
__global__ __launch_bounds__(256) void wtrans_bf16_kernel(const float* __restrict__ W,
                                                          u16* __restrict__ Wt,
                                                          int K, int N) {
    int idx = blockIdx.x * 256 + threadIdx.x;     // over N*K
    int nr = idx / K, kk = idx - nr * K;
    Wt[idx] = f2bfu(W[(size_t)kk * N + nr]);
}

// ---------------------------------------------------------------------------
// Causal local-window (L=5) softmax pooling, bf16 output. One wave per token.
// ---------------------------------------------------------------------------
__global__ __launch_bounds__(256) void pool_kernel(const float* __restrict__ x,
                                                   u16* __restrict__ y,
                                                   int S, int D) {
    const int lane = threadIdx.x & 31;
    const int tok  = blockIdx.x * 8 + (threadIdx.x >> 5);
    const int b = tok / S, s = tok % S;
    const float* xb = x + (size_t)b * S * D;
    const int e0 = lane * 16;

    float self[16];
#pragma unroll
    for (int i = 0; i < 16; ++i) self[i] = xb[(size_t)s * D + e0 + i];

    float wv[5][16], sc[5];
#pragma unroll
    for (int j = 0; j < 5; ++j) {
        int ss = s - 4 + j;
        float dot = 0.f;
        if (ss >= 0) {
#pragma unroll
            for (int i = 0; i < 16; ++i) {
                float f = xb[(size_t)ss * D + e0 + i];
                wv[j][i] = f;
                dot += f * self[i];
            }
        } else {
#pragma unroll
            for (int i = 0; i < 16; ++i) wv[j][i] = 0.f;
        }
#pragma unroll
        for (int off = 16; off >= 1; off >>= 1) dot += __shfl_xor(dot, off, 32);
        sc[j] = dot * 0.04419417382415922f;   // 1/sqrt(512)
    }
    float mx = sc[0];
#pragma unroll
    for (int j = 1; j < 5; ++j) mx = fmaxf(mx, sc[j]);
    float e[5], sum = 0.f;
#pragma unroll
    for (int j = 0; j < 5; ++j) { e[j] = __expf(sc[j] - mx); sum += e[j]; }
    float inv = 1.f / sum;

    float ov[16];
#pragma unroll
    for (int i = 0; i < 16; ++i) ov[i] = 0.f;
#pragma unroll
    for (int j = 0; j < 5; ++j) {
        float wj = e[j] * inv;
#pragma unroll
        for (int i = 0; i < 16; ++i) ov[i] += wj * wv[j][i];
    }
    u16* yp = y + (size_t)b * S * D + (size_t)s * D + e0;
#pragma unroll
    for (int i = 0; i < 8; ++i) *(u32*)(yp + 2 * i) = pack2(ov[2 * i], ov[2 * i + 1]);
}

// ---------------------------------------------------------------------------
// C = A[M,K](bf16) * Wt[N,K](bf16)^T + bias. 8 waves; 128x64 tile; K step 32.
// TDM double-buffered tile streaming. mode 0: f32 C[M][N];
// mode 2: bf16 Vt[(b*H+h)*64+dd][S] (per-head transposed V for attention).
// ---------------------------------------------------------------------------
__global__ __launch_bounds__(256) void gemm_bf16_kernel(
    const u16* __restrict__ A, const u16* __restrict__ Wt,
    const float* __restrict__ bias, void* __restrict__ Cout,
    int M, int K, int N, int mode, int S, int H) {
    __shared__ u16 As[2][128][32];
    __shared__ u16 Bs[2][64][32];

    const int t = threadIdx.x, lane = t & 31, w = t >> 5;
    const int hi = (lane >> 4) & 1, ln = lane & 15;
    const int wgM = blockIdx.y * 128, wgN = blockIdx.x * 64;
    const int T = K >> 5;

    v8f acc[4];
#pragma unroll
    for (int n = 0; n < 4; ++n) acc[n] = (v8f){};

    const u16* Ag = A + (size_t)wgM * K;
    const u16* Wg = Wt + (size_t)wgN * K;

#if USE_TDM
    if (t < 32) {
        tdm_load_2d(Ag, (u32)(size_t)&As[0][0][0], 32, 128, K);
        tdm_load_2d(Wg, (u32)(size_t)&Bs[0][0][0], 32, 64, K);
    }
#else
    coop_copy_2d(Ag, &As[0][0][0], 32, 128, K, t);
    coop_copy_2d(Wg, &Bs[0][0][0], 32, 64, K, t);
#endif

    for (int it = 0; it < T; ++it) {
        const int cur = it & 1, nxt = cur ^ 1;
#if USE_TDM
        if (t < 32) {
            if (it + 1 < T) {
                tdm_load_2d(Ag + (it + 1) * 32, (u32)(size_t)&As[nxt][0][0], 32, 128, K);
                tdm_load_2d(Wg + (it + 1) * 32, (u32)(size_t)&Bs[nxt][0][0], 32, 64, K);
                __builtin_amdgcn_s_wait_tensorcnt((unsigned short)2);  // tile `it` done
            } else {
                __builtin_amdgcn_s_wait_tensorcnt((unsigned short)0);
            }
        }
#else
        if (it + 1 < T) {
            coop_copy_2d(Ag + (it + 1) * 32, &As[nxt][0][0], 32, 128, K, t);
            coop_copy_2d(Wg + (it + 1) * 32, &Bs[nxt][0][0], 32, 64, K, t);
        }
#endif
        __syncthreads();
        FragBF a;
        ldsFrag(a, &As[cur][w * 16 + ln][0], hi);
#pragma unroll
        for (int n = 0; n < 4; ++n) {
            FragBF bfr;
            ldsFrag(bfr, &Bs[cur][n * 16 + ln][0], hi);
            acc[n] = __builtin_amdgcn_wmma_f32_16x16x32_bf16(
                false, a.v, false, bfr.v, (short)0, acc[n], false, false);
        }
        __syncthreads();
    }

    if (mode == 0) {
        float* Cf = (float*)Cout;
#pragma unroll
        for (int n = 0; n < 4; ++n) {
            int col = wgN + n * 16 + ln;
            float bv = bias[col];
#pragma unroll
            for (int r = 0; r < 8; ++r) {
                int row = wgM + w * 16 + hi * 8 + r;
                Cf[(size_t)row * N + col] = acc[n][r] + bv;
            }
        }
    } else {   // mode 2: per-head transposed bf16 V
        u16* Ct = (u16*)Cout;
#pragma unroll
        for (int n = 0; n < 4; ++n) {
            int col = wgN + n * 16 + ln;
            float bv = bias[col];
            int hh = col >> 6, dd = col & 63;
#pragma unroll
            for (int r = 0; r < 8; ++r) {
                int row = wgM + w * 16 + hi * 8 + r;       // = b*S + s
                int bb = row / S, ss = row - bb * S;
                Ct[(((size_t)bb * H + hh) * 64 + dd) * S + ss] = f2bfu(acc[n][r] + bv);
            }
        }
    }
}

// ---------------------------------------------------------------------------
// Flash attention per (b,h). q,k: bf16 [tok][D]; Vt: bf16 [(b*H+h)*64+dd][S].
// 8 waves x 16 queries; 32-key K/Vt tiles streamed by TDM, double-buffered.
// Output: bf16 x [tok][D] (feeds final GEMM).
// ---------------------------------------------------------------------------
__global__ __launch_bounds__(256) void attn_kernel(
    const u16* __restrict__ q, const u16* __restrict__ k,
    const u16* __restrict__ Vt, u16* __restrict__ out,
    int B, int S, int D, int H) {
    __shared__ u16 Ks[2][32][64];    // [key][dk]
    __shared__ u16 Vs[2][64][32];    // [dk][key]
    __shared__ u16 Ps[8][16][32];    // per-wave P transpose buffer

    const int t = threadIdx.x, lane = t & 31, w = t >> 5;
    const int hi = (lane >> 4) & 1, ln = lane & 15;
    const int bh = blockIdx.y;
    const int b = bh / H, h = bh - b * H;
    const int q0 = blockIdx.x * 128;
    const size_t base = ((size_t)b * S) * D + (size_t)h * 64;
    const u16* Kg = k + base;                                   // stride D per key
    const u16* Vg = Vt + ((size_t)(b * H + h) * 64) * S;        // stride S per dim

    // q fragments (2 K-chunks over dk=64), dword pair loads from bf16 global
    FragBF qa[2];
    {
        int row = q0 + w * 16 + ln;
        const u16* qp = q + base + (size_t)row * D;
#pragma unroll
        for (int c = 0; c < 2; ++c)
#pragma unroll
            for (int j = 0; j < 8; ++j) {
                int jj = ((j < 4) ? j : (j + 4)) + hi * 4;
                qa[c].d[j] = *(const u32*)(qp + 32 * c + 2 * jj);
            }
    }

    v8f o[4];
#pragma unroll
    for (int n = 0; n < 4; ++n) o[n] = (v8f){};
    float m[8], l[8];
#pragma unroll
    for (int r = 0; r < 8; ++r) { m[r] = -__builtin_inff(); l[r] = 0.f; }
    const float scale = 0.125f;   // 1/sqrt(64)
    const int T = S >> 5;

#if USE_TDM
    if (t < 32) {
        tdm_load_2d(Kg, (u32)(size_t)&Ks[0][0][0], 64, 32, D);
        tdm_load_2d(Vg, (u32)(size_t)&Vs[0][0][0], 32, 64, S);
    }
#else
    coop_copy_2d(Kg, &Ks[0][0][0], 64, 32, D, t);
    coop_copy_2d(Vg, &Vs[0][0][0], 32, 64, S, t);
#endif

    for (int it = 0; it < T; ++it) {
        const int cur = it & 1, nxt = cur ^ 1;
#if USE_TDM
        if (t < 32) {
            if (it + 1 < T) {
                tdm_load_2d(Kg + (size_t)(it + 1) * 32 * D, (u32)(size_t)&Ks[nxt][0][0], 64, 32, D);
                tdm_load_2d(Vg + (it + 1) * 32,             (u32)(size_t)&Vs[nxt][0][0], 32, 64, S);
                __builtin_amdgcn_s_wait_tensorcnt((unsigned short)2);
            } else {
                __builtin_amdgcn_s_wait_tensorcnt((unsigned short)0);
            }
        }
#else
        if (it + 1 < T) {
            coop_copy_2d(Kg + (size_t)(it + 1) * 32 * D, &Ks[nxt][0][0], 64, 32, D, t);
            coop_copy_2d(Vg + (it + 1) * 32,             &Vs[nxt][0][0], 32, 64, S, t);
        }
#endif
        __syncthreads();

        // S = q . k^T (two 16-key subtiles, dk accumulated over 2 chunks)
        v8f s0 = (v8f){}, s1 = (v8f){};
#pragma unroll
        for (int c = 0; c < 2; ++c) {
            FragBF b0, b1;
            ldsFrag(b0, &Ks[cur][ln][32 * c], hi);
            ldsFrag(b1, &Ks[cur][16 + ln][32 * c], hi);
            s0 = __builtin_amdgcn_wmma_f32_16x16x32_bf16(
                false, qa[c].v, false, b0.v, (short)0, s0, false, false);
            s1 = __builtin_amdgcn_wmma_f32_16x16x32_bf16(
                false, qa[c].v, false, b1.v, (short)0, s1, false, false);
        }

        // online softmax; rows map to (vgpr r, lane-half) -> reduce xor{1,2,4,8}
        float alpha[8], p0[8], p1[8];
#pragma unroll
        for (int r = 0; r < 8; ++r) {
            float a0 = s0[r] * scale, a1 = s1[r] * scale;
            float tmax = fmaxf(a0, a1);
#pragma unroll
            for (int off = 8; off >= 1; off >>= 1)
                tmax = fmaxf(tmax, __shfl_xor(tmax, off, 32));
            float mn = fmaxf(m[r], tmax);
            alpha[r] = __expf(m[r] - mn);
            p0[r] = __expf(a0 - mn);
            p1[r] = __expf(a1 - mn);
            float rs = p0[r] + p1[r];
#pragma unroll
            for (int off = 8; off >= 1; off >>= 1)
                rs += __shfl_xor(rs, off, 32);
            l[r] = alpha[r] * l[r] + rs;
            m[r] = mn;
        }

        // transpose P (16x32) through per-wave LDS into A-fragment layout
#pragma unroll
        for (int r = 0; r < 8; ++r) {
            Ps[w][hi * 8 + r][ln]      = f2bfu(p0[r]);
            Ps[w][hi * 8 + r][16 + ln] = f2bfu(p1[r]);
        }
        asm volatile("s_wait_dscnt 0" ::: "memory");   // wave-local LDS visibility
        FragBF pa;
        ldsFrag(pa, &Ps[w][ln][0], hi);

        // O = alpha*O + P x V
#pragma unroll
        for (int n = 0; n < 4; ++n) {
#pragma unroll
            for (int r = 0; r < 8; ++r) o[n][r] *= alpha[r];
            FragBF bv;
            ldsFrag(bv, &Vs[cur][n * 16 + ln][0], hi);
            o[n] = __builtin_amdgcn_wmma_f32_16x16x32_bf16(
                false, pa.v, false, bv.v, (short)0, o[n], false, false);
        }
        __syncthreads();
    }

#pragma unroll
    for (int n = 0; n < 4; ++n)
#pragma unroll
        for (int r = 0; r < 8; ++r) {
            int row = q0 + w * 16 + hi * 8 + r;
            out[base + (size_t)row * D + n * 16 + ln] = f2bfu(o[n][r] / l[r]);
        }
}

// ---------------------------------------------------------------------------
extern "C" void kernel_launch(void* const* d_in, const int* in_sizes, int n_in,
                              void* d_out, int out_size, void* d_ws, size_t ws_size,
                              hipStream_t stream) {
    const float* query = (const float*)d_in[0];
    const float* key   = (const float*)d_in[1];
    const float* value = (const float*)d_in[2];
    /* d_in[3] = mask: unused by the reference */
    const float* W_fk  = (const float*)d_in[4];
    const float* b_fk  = (const float*)d_in[5];
    const float* W0    = (const float*)d_in[6];
    const float* b0    = (const float*)d_in[7];
    const float* Wout  = (const float*)d_in[8];
    const float* bout  = (const float*)d_in[9];
    float* out = (float*)d_out;

    const int B = 4, S = 2048, D = 512, H = 8;
    const int NT = B * S;
    const size_t NTD = (size_t)NT * D;
    const size_t DD = (size_t)D * D;

    u16* key_bf = (u16*)d_ws;
    u16* val_bf = key_bf + NTD;
    u16* qbf    = val_bf + NTD;
    u16* kbf    = qbf + NTD;
    u16* xbf    = kbf + NTD;
    u16* Vt     = xbf + NTD;
    u16* Wfkt   = Vt + NTD;
    u16* W0t    = Wfkt + DD;
    u16* Woutt  = W0t + DD;
    float* fk   = (float*)(Woutt + DD);

    dim3 blk(256);
    f32_to_bf16_kernel<<<dim3(NTD / 1024), blk, 0, stream>>>(key, key_bf, (int)NTD);
    f32_to_bf16_kernel<<<dim3(NTD / 1024), blk, 0, stream>>>(value, val_bf, (int)NTD);
    wtrans_bf16_kernel<<<dim3(DD / 256), blk, 0, stream>>>(W_fk, Wfkt, D, D);
    wtrans_bf16_kernel<<<dim3(DD / 256), blk, 0, stream>>>(W0, W0t, D, D);
    wtrans_bf16_kernel<<<dim3(DD / 256), blk, 0, stream>>>(Wout, Woutt, D, D);

    pool_kernel<<<dim3(NT / 8), blk, 0, stream>>>(query, qbf, S, D);
    gemm_bf16_kernel<<<dim3(D / 64, NT / 128), blk, 0, stream>>>(
        key_bf, Wfkt, b_fk, fk, NT, D, D, 0, S, H);
    pool_kernel<<<dim3(NT / 8), blk, 0, stream>>>(fk, kbf, S, D);
    gemm_bf16_kernel<<<dim3(D / 64, NT / 128), blk, 0, stream>>>(
        val_bf, W0t, b0, Vt, NT, D, D, 2, S, H);
    attn_kernel<<<dim3(S / 128, B * H), blk, 0, stream>>>(qbf, kbf, Vt, xbf, B, S, D, H);
    gemm_bf16_kernel<<<dim3(D / 64, NT / 128), blk, 0, stream>>>(
        xbf, Woutt, bout, out, NT, D, D, 0, S, H);
}
